// LDGATv1_5789615915614
// MI455X (gfx1250) — matrix operand order repaired and checked by reading, not compiled, not run on hardware
//
#include <hip/hip_runtime.h>
#include <hip/hip_bf16.h>

// ---------------------------------------------------------------------------
// LDGATv1 forward on MI455X (gfx1250, wave32).
// Dense GEMMs: LDS-staged v_wmma_f32_16x16x32_f16 (f32 accumulate),
// 4 N-subtiles per wave, A fragment reused, WMMAs issued back-to-back.
// Irregular graph work (KNN top-30, edge softmax) in LDS-resident kernels.
// ---------------------------------------------------------------------------

#define NTOT 8192      // B*P
#define PPC  2048      // points per cloud
#define KNBR 30
#define KE   31        // neighbors + self-loop
#define NHEAD 3

#define MT 128         // GEMM M tile (8 waves x 16)
#define NTW 64         // GEMM N tile per block (4 x 16 subtiles per wave)
#define NSUB 4
#define KT 32          // GEMM K chunk (one WMMA depth)
#define AS_STRIDE 40   // LDS row stride (halfs): 80B -> conflict-free b128 reads
#define BS_STRIDE 40

typedef __attribute__((ext_vector_type(16))) _Float16 v16h;
typedef __attribute__((ext_vector_type(8)))  _Float16 v8h;
typedef __attribute__((ext_vector_type(8)))  float    v8f;
typedef __attribute__((ext_vector_type(2)))  float    f2;

// ---------------------------------------------------------------------------
// WMMA GEMM: C[M,Nc] = act(A[M,K] @ B[K,Nc] + bias)
// A row-major, row stride lda (8B-aligned rows). B row-major packed [K,Nc]
// (Nc even). C row stride ldc. Block = 256 threads = 8 wave32s; wave w owns
// rows [blockIdx.y*128 + w*16, +16) across 4 column subtiles of 16 at
// blockIdx.x*64. M must be a multiple of 128 (always 8192 here).
// Register layouts per CDNA5 ISA 7.12.2.
// ---------------------------------------------------------------------------
__global__ __launch_bounds__(256)
void wmma_gemm_kernel(const float* __restrict__ A, int lda,
                      const float* __restrict__ B,
                      const float* __restrict__ bias,
                      float* __restrict__ C, int ldc,
                      int M, int K, int Nc, int relu)
{
  __shared__ _Float16 As[MT * AS_STRIDE];    // [row][k] f16, padded rows
  __shared__ _Float16 Bst[NTW * BS_STRIDE];  // transposed: [col][k] f16

  const int tid  = threadIdx.x;
  const int lane = tid & 31;
  const int wave = tid >> 5;
  const int hs   = lane >> 4;     // lane half: 0 (lanes 0-15), 1 (16-31)
  const int lid  = lane & 15;

  const int tile_n = blockIdx.x * NTW;
  const int tile_m = blockIdx.y * MT;

  v8f acc0 = {}, acc1 = {}, acc2 = {}, acc3 = {};

  for (int k0 = 0; k0 < K; k0 += KT) {
    __syncthreads();   // previous chunk's LDS reads done before overwrite

    // ---- stage A panel: 128 x 32 fp32 -> f16 LDS (float2 granularity) ----
    if (k0 + KT <= K) {
      // phase 1: issue all 8 loads (clause-friendly) + unconditional
      // speculative prefetch of the next K chunk (dropped if OOB).
      f2 av[8];
#pragma unroll
      for (int it = 0; it < 8; ++it) {
        int slot = tid + it * 256;           // 2048 slots of 2 floats
        int row  = slot >> 4;                // 128 rows
        int kq   = (slot & 15) << 1;         // 0,2,..,30
        const float* p = A + (size_t)(tile_m + row) * lda + k0 + kq;
        av[it] = *(const f2*)p;
        __builtin_prefetch(p + KT, 0, 1);
      }
      // phase 2: convert + LDS store
#pragma unroll
      for (int it = 0; it < 8; ++it) {
        int slot = tid + it * 256;
        int row  = slot >> 4;
        int kq   = (slot & 15) << 1;
        As[row * AS_STRIDE + kq    ] = (_Float16)av[it][0];
        As[row * AS_STRIDE + kq + 1] = (_Float16)av[it][1];
      }
    } else {
      // K tail: guarded scalar loads, zero-pad
#pragma unroll
      for (int it = 0; it < 8; ++it) {
        int slot = tid + it * 256;
        int row  = slot >> 4;
        int kq   = (slot & 15) << 1;
        int k    = k0 + kq;
        const float* Arow = A + (size_t)(tile_m + row) * lda;
        float x0 = (k     < K) ? Arow[k    ] : 0.0f;
        float x1 = (k + 1 < K) ? Arow[k + 1] : 0.0f;
        As[row * AS_STRIDE + kq    ] = (_Float16)x0;
        As[row * AS_STRIDE + kq + 1] = (_Float16)x1;
      }
    }

    // ---- stage B panel transposed: Bst[col][k], 32 k x 64 cols ----
    if (k0 + KT <= K && tile_n + NTW <= Nc) {
      // fast path: float2 over columns (Nc even, tile_n multiple of 64)
      f2 bv[4];
#pragma unroll
      for (int it = 0; it < 4; ++it) {
        int slot = tid + it * 256;           // 1024 slots: 32 k x 32 col-pairs
        int kk   = slot >> 5;
        int cp   = (slot & 31) << 1;
        bv[it] = *(const f2*)(B + (size_t)(k0 + kk) * Nc + tile_n + cp);
      }
#pragma unroll
      for (int it = 0; it < 4; ++it) {
        int slot = tid + it * 256;
        int kk   = slot >> 5;
        int cp   = (slot & 31) << 1;
        Bst[(cp    ) * BS_STRIDE + kk] = (_Float16)bv[it][0];
        Bst[(cp + 1) * BS_STRIDE + kk] = (_Float16)bv[it][1];
      }
    } else {
      // edge: guarded scalar loads, zero-pad
#pragma unroll
      for (int it = 0; it < 8; ++it) {
        int e  = tid + it * 256;             // 2048 elements (32 k x 64 col)
        int kk = e >> 6;
        int c  = e & 63;
        int gk = k0 + kk;
        int gc = tile_n + c;
        float v = (gk < K && gc < Nc) ? B[(size_t)gk * Nc + gc] : 0.0f;
        Bst[c * BS_STRIDE + kk] = (_Float16)v;
      }
    }
    __syncthreads();

    // ---- A fragment (2x ds_load_b128), reused by all 4 WMMAs ----
    const int arow = wave * 16 + lid;   // A row within tile (== D row set)
    v8h alo = *(const v8h*)&As[arow * AS_STRIDE + hs * 8];        // K = hs*8+0..7
    v8h ahi = *(const v8h*)&As[arow * AS_STRIDE + 16 + hs * 8];   // K = 16+hs*8+0..7
    v16h a;
#pragma unroll
    for (int i = 0; i < 8; ++i) { a[i] = alo[i]; a[8 + i] = ahi[i]; }

    // ---- load all 4 B fragments first (one dscnt wait), then 4 WMMAs ----
    v8h bf[2 * NSUB];
#pragma unroll
    for (int j = 0; j < NSUB; ++j) {
      const _Float16* bp = &Bst[(j * 16 + lid) * BS_STRIDE + hs * 8];
      bf[2 * j    ] = *(const v8h*)bp;          // K = hs*8 + 0..7
      bf[2 * j + 1] = *(const v8h*)(bp + 16);   // K = 16 + hs*8 + 0..7
    }
#pragma unroll
    for (int j = 0; j < NSUB; ++j) {
      v16h b;
#pragma unroll
      for (int i = 0; i < 8; ++i) { b[i] = bf[2 * j][i]; b[8 + i] = bf[2 * j + 1][i]; }
      v8f c = (j == 0) ? acc0 : (j == 1) ? acc1 : (j == 2) ? acc2 : acc3;
      c = __builtin_amdgcn_wmma_f32_16x16x32_f16(
          /*neg_a=*/false, a, /*neg_b=*/false, b,
          /*c_mod=*/(short)0, c, /*reuse_a=*/false, /*reuse_b=*/false);
      if      (j == 0) acc0 = c;
      else if (j == 1) acc1 = c;
      else if (j == 2) acc2 = c;
      else             acc3 = c;
    }
  }

  // ---- epilogue: D VGPR r -> row = tile_m + wave*16 + hs*8 + r ----
#pragma unroll
  for (int j = 0; j < NSUB; ++j) {
    int col = tile_n + j * 16 + lid;
    if (col < Nc) {
      float bv = bias ? bias[col] : 0.0f;
      v8f acc = (j == 0) ? acc0 : (j == 1) ? acc1 : (j == 2) ? acc2 : acc3;
#pragma unroll
      for (int r = 0; r < 8; ++r) {
        int row = tile_m + wave * 16 + (hs << 3) + r;
        if (row < M) {
          float v = acc[r] + bv;
          if (relu) v = fmaxf(v, 0.0f);
          C[(size_t)row * ldc + col] = v;
        }
      }
    }
  }
}

// ---------------------------------------------------------------------------
// x0 = concat(x, pos) into link[:, 0:6] (link row stride 326)
// ---------------------------------------------------------------------------
__global__ void x0_kernel(const float* __restrict__ x, const float* __restrict__ pos,
                          float* __restrict__ link)
{
  int n = blockIdx.x * blockDim.x + threadIdx.x;
  if (n >= NTOT) return;
  float* row = link + (size_t)n * 326;
#pragma unroll
  for (int c = 0; c < 3; ++c) {
    row[c]     = x[n * 3 + c];
    row[3 + c] = pos[n * 3 + c];
  }
}

// ---------------------------------------------------------------------------
// Exact per-cloud KNN (K=30, no self), self-loop appended at slot 30.
// One block (256 threads) per query point. Distances in LDS, then 30
// deterministic block-argmin extractions.
// ---------------------------------------------------------------------------
__global__ __launch_bounds__(256)
void knn_kernel(const float* __restrict__ feat, int lda, int C,
                int* __restrict__ nbr)
{
  __shared__ float q[64];
  __shared__ float dist[PPC];
  __shared__ float rmin[256];
  __shared__ int   ridx[256];

  const int n   = blockIdx.x;
  const int bc  = n / PPC;          // cloud id
  const int tid = threadIdx.x;

  for (int c = tid; c < C; c += 256) q[c] = feat[(size_t)n * lda + c];
  __syncthreads();

  const int C2 = C >> 1;            // C is 6 or 64 (always even)
  for (int j = tid; j < PPC; j += 256) {
    int cand = bc * PPC + j;
    const float* fr = feat + (size_t)cand * lda;
    float d = 0.0f;
    for (int c = 0; c < C2; ++c) {
      f2 v = *(const f2*)(fr + 2 * c);
      float d0 = v[0] - q[2 * c];
      float d1 = v[1] - q[2 * c + 1];
      d += d0 * d0 + d1 * d1;
    }
    dist[j] = (cand == n) ? 3.0e38f : d;   // exclude self
  }
  __syncthreads();

  for (int k = 0; k < KNBR; ++k) {
    float best = 3.0e38f; int bi = 0;
    for (int j = tid; j < PPC; j += 256) {
      float d = dist[j];
      if (d < best) { best = d; bi = j; }
    }
    rmin[tid] = best; ridx[tid] = bi;
    __syncthreads();
    for (int s = 128; s > 0; s >>= 1) {
      if (tid < s) {
        if (rmin[tid + s] < rmin[tid]) { rmin[tid] = rmin[tid + s]; ridx[tid] = ridx[tid + s]; }
      }
      __syncthreads();
    }
    if (tid == 0) {
      nbr[(size_t)n * KE + k] = bc * PPC + ridx[0];
      dist[ridx[0]] = 3.0e38f;
    }
    __syncthreads();
  }
  if (tid == 0) nbr[(size_t)n * KE + KNBR] = n;   // self-loop last
}

// ---------------------------------------------------------------------------
// Per-(node,head) attention scores: s = <xh, a_src>, t = <xh, a_dst>
// ---------------------------------------------------------------------------
__global__ void st_kernel(const float* __restrict__ xh,
                          const float* __restrict__ a_src,
                          const float* __restrict__ a_dst,
                          float* __restrict__ s, float* __restrict__ t, int F)
{
  int i = blockIdx.x * blockDim.x + threadIdx.x;   // over NTOT*NHEAD
  if (i >= NTOT * NHEAD) return;
  int n = i / NHEAD, h = i % NHEAD;
  const float* row = xh + (size_t)n * NHEAD * F + (size_t)h * F;
  const float* as  = a_src + (size_t)h * F;
  const float* ad  = a_dst + (size_t)h * F;
  float ss = 0.0f, tt = 0.0f;
  for (int f = 0; f < F; ++f) { ss += row[f] * as[f]; tt += row[f] * ad[f]; }
  s[i] = ss; t[i] = tt;
}

// ---------------------------------------------------------------------------
// GAT edge-softmax + aggregation. One block per node.
// e[k,h] = leaky_relu(s[src] + t[n], 0.2); alpha = softmax over k (31);
// out[n,h,f] = sum_k alpha * xh[src,h,f] + bias.
// ---------------------------------------------------------------------------
__global__ __launch_bounds__(256)
void gat_kernel(const float* __restrict__ xh,
                const float* __restrict__ s, const float* __restrict__ t,
                const int* __restrict__ nbr, const float* __restrict__ bias,
                float* __restrict__ out, int F)
{
  __shared__ int   sn[KE];
  __shared__ float al[KE * NHEAD];
  __shared__ float sinv[NHEAD];

  const int n   = blockIdx.x;
  const int tid = threadIdx.x;

  if (tid < KE) sn[tid] = nbr[(size_t)n * KE + tid];
  __syncthreads();

  if (tid < KE * NHEAD) {
    int k = tid / NHEAD, h = tid % NHEAD;
    float e = s[(size_t)sn[k] * NHEAD + h] + t[(size_t)n * NHEAD + h];
    al[k * NHEAD + h] = (e > 0.0f) ? e : 0.2f * e;   // leaky relu 0.2
  }
  __syncthreads();

  if (tid < NHEAD) {
    float m = -3.0e38f;
    for (int k = 0; k < KE; ++k) m = fmaxf(m, al[k * NHEAD + tid]);
    float sum = 0.0f;
    for (int k = 0; k < KE; ++k) {
      float ex = __expf(al[k * NHEAD + tid] - m);
      al[k * NHEAD + tid] = ex;
      sum += ex;
    }
    sinv[tid] = 1.0f / sum;
  }
  __syncthreads();

  const int HF = NHEAD * F;
  for (int o = tid; o < HF; o += 256) {
    int h = o / F;
    float inv = sinv[h];
    float acc = 0.0f;
    for (int k = 0; k < KE; ++k)
      acc += al[k * NHEAD + h] * xh[(size_t)sn[k] * HF + o];
    out[(size_t)n * HF + o] = acc * inv + bias[o];
  }
}

// ---------------------------------------------------------------------------
// Column-wise max over all N rows (global feature), 1024 columns.
// ---------------------------------------------------------------------------
__global__ void colmax_kernel(const float* __restrict__ h, float* __restrict__ g)
{
  int j = blockIdx.x * blockDim.x + threadIdx.x;
  if (j >= 1024) return;
  float m = -3.0e38f;
  for (int r = 0; r < NTOT; ++r) m = fmaxf(m, h[(size_t)r * 1024 + j]);
  g[j] = m;
}

// gvec[j] = mb1[j] + sum_k g[k] * mw1[326+k, j]   (fold global feature into bias)
__global__ void gvec_kernel(const float* __restrict__ g,
                            const float* __restrict__ mw1,
                            const float* __restrict__ mb1,
                            float* __restrict__ gv)
{
  int j = blockIdx.x * blockDim.x + threadIdx.x;
  if (j >= 256) return;
  float acc = mb1[j];
  for (int k = 0; k < 1024; ++k) acc += g[k] * mw1[(size_t)(326 + k) * 256 + j];
  gv[j] = acc;
}

// In-place log_softmax over the 50-way class dim.
__global__ void lsm_kernel(float* __restrict__ out)
{
  int n = blockIdx.x * blockDim.x + threadIdx.x;
  if (n >= NTOT) return;
  float* r = out + (size_t)n * 50;
  float m = -3.0e38f;
  for (int c = 0; c < 50; ++c) m = fmaxf(m, r[c]);
  float sum = 0.0f;
  for (int c = 0; c < 50; ++c) sum += __expf(r[c] - m);
  float ls = logf(sum);
  for (int c = 0; c < 50; ++c) r[c] = r[c] - m - ls;
}

// ---------------------------------------------------------------------------
// Host orchestration
// ---------------------------------------------------------------------------
extern "C" void kernel_launch(void* const* d_in, const int* in_sizes, int n_in,
                              void* d_out, int out_size, void* d_ws, size_t ws_size,
                              hipStream_t stream) {
  (void)in_sizes; (void)n_in; (void)out_size; (void)ws_size;

  const float* x    = (const float*)d_in[0];
  const float* pos  = (const float*)d_in[1];
  /* d_in[2] = batch (unused: cloud id == n / PPC) */
  const float* w1   = (const float*)d_in[3];
  const float* as1  = (const float*)d_in[4];
  const float* ad1  = (const float*)d_in[5];
  const float* b1   = (const float*)d_in[6];
  const float* ml1w = (const float*)d_in[7];
  const float* ml1b = (const float*)d_in[8];
  const float* w2   = (const float*)d_in[9];
  const float* as2  = (const float*)d_in[10];
  const float* ad2  = (const float*)d_in[11];
  const float* b2   = (const float*)d_in[12];
  const float* ml2w = (const float*)d_in[13];
  const float* ml2b = (const float*)d_in[14];
  const float* w3   = (const float*)d_in[15];
  const float* as3  = (const float*)d_in[16];
  const float* ad3  = (const float*)d_in[17];
  const float* b3   = (const float*)d_in[18];
  const float* ml3w = (const float*)d_in[19];
  const float* ml3b = (const float*)d_in[20];
  const float* w4   = (const float*)d_in[21];
  const float* as4  = (const float*)d_in[22];
  const float* ad4  = (const float*)d_in[23];
  const float* b4   = (const float*)d_in[24];
  const float* ml4w = (const float*)d_in[25];
  const float* ml4b = (const float*)d_in[26];
  const float* few1 = (const float*)d_in[27];
  const float* feb1 = (const float*)d_in[28];
  const float* few2 = (const float*)d_in[29];
  const float* feb2 = (const float*)d_in[30];
  const float* mw1  = (const float*)d_in[31];
  const float* mb1  = (const float*)d_in[32];
  const float* mw2  = (const float*)d_in[33];
  const float* mb2  = (const float*)d_in[34];
  const float* mw3  = (const float*)d_in[35];
  const float* mb3  = (const float*)d_in[36];
  const float* mw4  = (const float*)d_in[37];
  const float* mb4  = (const float*)d_in[38];

  float* out = (float*)d_out;   // [NTOT, 50]

  // ---- bump allocator over d_ws (deterministic layout) ----
  char*  wsp = (char*)d_ws;
  size_t off = 0;
  auto alloc = [&](size_t bytes) -> char* {
    char* p = wsp + off;
    off += (bytes + 255) & ~(size_t)255;
    return p;
  };
  float* link = (float*)alloc((size_t)NTOT * 326 * 4);  // [x0|x1|x2|x3|x4]
  int*   nbr  = (int*)  alloc((size_t)NTOT * KE * 4);
  float* xh   = (float*)alloc((size_t)NTOT * 384 * 4);  // per-layer xf@W
  float* sbuf = (float*)alloc((size_t)NTOT * NHEAD * 4);
  float* tbuf = (float*)alloc((size_t)NTOT * NHEAD * 4);
  float* gout = (float*)alloc((size_t)NTOT * 384 * 4);  // GAT output
  float* h1   = (float*)alloc((size_t)NTOT * 1024 * 4);
  float* h2   = (float*)alloc((size_t)NTOT * 1024 * 4);
  float* g    = (float*)alloc(1024 * 4);
  float* gv   = (float*)alloc(256 * 4);
  // head buffers alias retired GAT scratch
  float* hh1  = xh;                                       // [NTOT,256]
  float* hh2  = gout;                                     // [NTOT,256]
  float* hh3  = (float*)((char*)xh + (size_t)NTOT * 256 * 4); // [NTOT,128]

  auto gemm = [&](const float* A, int lda, const float* B, const float* bias,
                  float* C, int ldc, int K, int Nc, int relu) {
    dim3 grid((Nc + NTW - 1) / NTW, (NTOT + MT - 1) / MT);
    wmma_gemm_kernel<<<grid, 256, 0, stream>>>(A, lda, B, bias, C, ldc,
                                               NTOT, K, Nc, relu);
  };

  // ---- x0 = [x, pos] ----
  x0_kernel<<<(NTOT + 255) / 256, 256, 0, stream>>>(x, pos, link);

  // ---- GAT layer descriptors ----
  struct Layer {
    const float *W, *asrc, *adst, *bias, *mlw, *mlb;
    int Kin, F, coloff, knn_col, knn_C, do_knn;
  };
  const Layer layers[4] = {
    { w1, as1, ad1, b1, ml1w, ml1b,   6,  64,   6,   0,  6, 1 },
    { w2, as2, ad2, b2, ml2w, ml2b,  70,  64,  70,   6, 64, 1 },
    { w3, as3, ad3, b3, ml3w, ml3b, 134,  64, 134,  70, 64, 1 },
    { w4, as4, ad4, b4, ml4w, ml4b, 198, 128, 198,  70, 64, 0 }, // reuses knn(x2)
  };

  for (int l = 0; l < 4; ++l) {
    const Layer& L = layers[l];
    const int HF = NHEAD * L.F;
    if (L.do_knn)
      knn_kernel<<<NTOT, 256, 0, stream>>>(link + L.knn_col, 326, L.knn_C, nbr);
    // xh = xf @ W   (no bias, no act)
    gemm(link, 326, L.W, nullptr, xh, HF, L.Kin, HF, 0);
    // attention scores
    st_kernel<<<(NTOT * NHEAD + 255) / 256, 256, 0, stream>>>(xh, L.asrc, L.adst,
                                                              sbuf, tbuf, L.F);
    // edge softmax + aggregate (+ GAT bias)
    gat_kernel<<<NTOT, 256, 0, stream>>>(xh, sbuf, tbuf, nbr, L.bias, gout, L.F);
    // x_l = gat @ mlw + mlb  -> written into link column slice
    gemm(gout, HF, L.mlw, L.mlb, link + L.coloff, 326, HF, L.F, 0);
  }

  // ---- fe_mlp: h2 = relu(link@few1+feb1) @ few2 + feb2 ----
  gemm(link, 326, few1, feb1, h1, 1024, 326, 1024, 1);
  gemm(h1, 1024, few2, feb2, h2, 1024, 1024, 1024, 0);

  // ---- global feature + fold into head bias ----
  colmax_kernel<<<(1024 + 255) / 256, 256, 0, stream>>>(h2, g);
  gvec_kernel<<<1, 256, 0, stream>>>(g, mw1, mb1, gv);

  // ---- head MLP ----
  gemm(link, 326, mw1, gv, hh1, 256, 326, 256, 1);   // mw1 rows 0..325 + folded g part
  gemm(hh1, 256, mw2, mb2, hh2, 256, 256, 256, 1);
  gemm(hh2, 256, mw3, mb3, hh3, 128, 256, 128, 1);
  gemm(hh3, 128, mw4, mb4, out, 50, 128, 50, 0);

  // ---- log_softmax in place ----
  lsm_kernel<<<(NTOT + 255) / 256, 256, 0, stream>>>(out);
}
